// CorrelatedCategoricalsLM_19078244729473
// MI455X (gfx1250) — matrix operand order repaired and checked by reading, not compiled, not run on hardware
//
#include <hip/hip_runtime.h>
#include <hip/hip_bf16.h>

typedef __bf16 bf16;
typedef __attribute__((ext_vector_type(16))) __bf16 v16bf;
typedef __attribute__((ext_vector_type(8)))  __bf16 v8bf;
typedef __attribute__((ext_vector_type(4)))  __bf16 v4bf;
typedef __attribute__((ext_vector_type(8)))  float  v8f;
typedef __attribute__((ext_vector_type(4)))  float  v4f;

#define B_    32
#define T_    128
#define V_    32000
#define E_    512
#define H_    512
#define DZ_   256
#define M_    (B_ * T_)     // 4096
#define KIN_  (E_ + DZ_)    // 768
#define G3H_  (3 * H_)      // 1536
#define SCAN_BLOCKS 16

// ---------------------------------------------------------------------------
// helpers
// ---------------------------------------------------------------------------
__device__ __forceinline__ float sig_fast(float x) {
    return 1.0f / (1.0f + __expf(-x));
}
__device__ __forceinline__ float tanh_fast(float x) {
    return 1.0f - 2.0f / (__expf(2.0f * x) + 1.0f);
}

__device__ __forceinline__ v8f wmma_bf16(v16bf a, v16bf b, v8f c) {
    // (neg_a, A, neg_b, B, c_mod, C, reuse_a, reuse_b)
    return __builtin_amdgcn_wmma_f32_16x16x32_bf16(
        false, a, false, b, (short)0, c, false, false);
}

// A fragment: 16x32 tile of row-major [rows][lda] bf16 matrix.
// Lane m=l%16, half=l/16; VGPR v<4: K = half*8 + 2v(+1); v>=4: K = 16+half*8+...
__device__ __forceinline__ v16bf load_frag_A(const bf16* __restrict__ A,
                                             int lda, int mBase, int kBase) {
    int lane = threadIdx.x & 31;
    int m = lane & 15, half = lane >> 4;
    const bf16* p = A + (size_t)(mBase + m) * lda + kBase;
    v8bf lo = *(const v8bf*)(p + half * 8);
    v8bf hi = *(const v8bf*)(p + 16 + half * 8);
    v16bf r;
#pragma unroll
    for (int i = 0; i < 8; i++) { r[i] = lo[i]; r[8 + i] = hi[i]; }
    return r;
}

// B fragment: 32x16 tile, weights stored [N][K] row-major (X @ W^T form).
// Lane n=l%16, half=l/16; elems 0..7: K = half*8 + 0..7; 8..15: K = 16+half*8+0..7.
__device__ __forceinline__ v16bf load_frag_B(const bf16* __restrict__ Bw,
                                             int ldb, int nBase, int kBase) {
    int lane = threadIdx.x & 31;
    int n = lane & 15, half = lane >> 4;
    const bf16* p = Bw + (size_t)(nBase + n) * ldb + kBase;
    v8bf lo = *(const v8bf*)(p + half * 8);
    v8bf hi = *(const v8bf*)(p + 16 + half * 8);
    v16bf r;
#pragma unroll
    for (int i = 0; i < 8; i++) { r[i] = lo[i]; r[8 + i] = hi[i]; }
    return r;
}

__device__ __forceinline__ void load_A2(v16bf (&a)[2], const bf16* __restrict__ A,
                                        int lda, int wm, int k) {
    a[0] = load_frag_A(A, lda, wm, k);
    a[1] = load_frag_A(A, lda, wm + 16, k);
}
__device__ __forceinline__ void load_B8(v16bf (&b)[8], const bf16* __restrict__ Bw,
                                        int ldb, int wn, int k) {
#pragma unroll
    for (int j = 0; j < 8; j++) b[j] = load_frag_B(Bw, ldb, wn + j * 16, k);
}
__device__ __forceinline__ void wmma_tile(v8f (&acc)[2][8], const v16bf (&a)[2],
                                          const v16bf (&b)[8]) {
#pragma unroll
    for (int i = 0; i < 2; i++)
#pragma unroll
        for (int j = 0; j < 8; j++)
            acc[i][j] = wmma_bf16(a[i], b[j], acc[i][j]);
}

// ---------------------------------------------------------------------------
// K1: fp32 -> bf16 convert, vectorized b128 NT loads (source streamed once)
// ---------------------------------------------------------------------------
__global__ void f32_to_bf16_v4(const float* __restrict__ in, bf16* __restrict__ out,
                               int n4) {
    int i = blockIdx.x * blockDim.x + threadIdx.x;
    int stride = gridDim.x * blockDim.x;
    for (; i < n4; i += stride) {
        v4f v = __builtin_nontemporal_load((const v4f*)in + i);
        v4bf o;
#pragma unroll
        for (int j = 0; j < 4; j++) o[j] = (bf16)v[j];
        *((v4bf*)out + i) = o;
    }
}

// ---------------------------------------------------------------------------
// K2: rnn_in = concat(emb[x], z) as bf16 [M][KIN], 8 elems/thread.
//     E=512 and KIN=768 are multiples of 8 -> a chunk never straddles emb/z.
// ---------------------------------------------------------------------------
__global__ void build_rnn_in(const int* __restrict__ x, const float* __restrict__ z,
                             const float* __restrict__ emb, bf16* __restrict__ out) {
    int idx = blockIdx.x * blockDim.x + threadIdx.x;      // chunk id
    if (idx >= M_ * (KIN_ / 8)) return;
    int bt = idx / (KIN_ / 8);
    int c8 = idx - bt * (KIN_ / 8);
    int c  = c8 * 8;
    const float* src;
    if (c < E_) src = emb + (size_t)x[bt] * E_ + c;
    else        src = z + (bt / T_) * DZ_ + (c - E_);
    v4f lo = *(const v4f*)src;
    v4f hi = *(const v4f*)(src + 4);
    v8bf o;
#pragma unroll
    for (int j = 0; j < 4; j++) { o[j] = (bf16)lo[j]; o[4 + j] = (bf16)hi[j]; }
    *(v8bf*)(out + (size_t)bt * KIN_ + c) = o;
}

// ---------------------------------------------------------------------------
// K3: h0 = tanh(z @ W_init^T + b_init); seeds bf16 h ping-pong buffer and
//     resets the scan's grid-barrier counter (stream order guarantees hb).
// ---------------------------------------------------------------------------
__global__ void h0_kernel(const float* __restrict__ z, const float* __restrict__ Wi,
                          const float* __restrict__ bi, float* __restrict__ h0f,
                          bf16* __restrict__ hbuf0, unsigned* __restrict__ bar) {
    int idx = blockIdx.x * blockDim.x + threadIdx.x;   // 0 .. 16383
    if (idx == 0)
        __hip_atomic_store(bar, 0u, __ATOMIC_RELAXED, __HIP_MEMORY_SCOPE_AGENT);
    int b = idx >> 9, n = idx & 511;
    const v4f* zr = (const v4f*)(z + b * DZ_);
    const v4f* wr = (const v4f*)(Wi + n * DZ_);
    float acc = bi[n];
    for (int k = 0; k < DZ_ / 4; k++) {
        v4f zv = zr[k], wv = wr[k];
#pragma unroll
        for (int j = 0; j < 4; j++) acc = fmaf(zv[j], wv[j], acc);
    }
    float h = tanh_fast(acc);
    h0f[idx] = h;
    hbuf0[idx] = (bf16)h;
}

// ---------------------------------------------------------------------------
// K4: GEMM  C[M][N] = A[M][K](bf16) x B[N][K](bf16)^T (+bias), fp32 out.
//     256 threads = 8 waves arranged 4(M) x 2(N); block tile 128x256,
//     wave tile 32x128 (16 accum tiles). 128-row M tiles halve W_out L2
//     re-read traffic vs 64 (the binding roofline: ~1.0 GB B + 0.5 GB A of
//     L2->WGP traffic vs ~27us of bf16 WMMA). K templated; double-buffered
//     fragments give a 2-deep pipeline: 20 b128 loads in flight per 16 WMMAs
//     (1.25 loads/WMMA). NT=true streams C non-temporally (keeps W_out in L2).
// ---------------------------------------------------------------------------
template <int K, bool BIAS, bool NT>
__global__ __launch_bounds__(256) void gemm_bf16_nt(
    const bf16* __restrict__ A, const bf16* __restrict__ Bw,
    const float* __restrict__ bias, float* __restrict__ C, int N) {
    int wave = threadIdx.x >> 5;
    int wm = blockIdx.y * 128 + (wave >> 1) * 32;   // 4 M-waves
    int wn = blockIdx.x * 256 + (wave & 1) * 128;   // 2 N-waves

    v8f acc[2][8];
#pragma unroll
    for (int i = 0; i < 2; i++)
#pragma unroll
        for (int j = 0; j < 8; j++)
            acc[i][j] = (v8f){0.f, 0.f, 0.f, 0.f, 0.f, 0.f, 0.f, 0.f};

    v16bf a0[2], a1[2], b0[8], b1[8];
    load_A2(a0, A, K, wm, 0);
    load_B8(b0, Bw, K, wn, 0);
    static_assert(K % 64 == 0, "pipeline assumes even k-step count");
#pragma unroll
    for (int k = 0; k < K; k += 64) {
        load_A2(a1, A, K, wm, k + 32);
        load_B8(b1, Bw, K, wn, k + 32);
        wmma_tile(acc, a0, b0);
        if (k + 64 < K) {
            load_A2(a0, A, K, wm, k + 64);
            load_B8(b0, Bw, K, wn, k + 64);
        }
        wmma_tile(acc, a1, b1);
    }

    int lane = threadIdx.x & 31;
    int n0 = lane & 15, half = lane >> 4;
#pragma unroll
    for (int i = 0; i < 2; i++)
#pragma unroll
        for (int j = 0; j < 8; j++) {
            int ncol = wn + j * 16 + n0;
            float bv = BIAS ? bias[ncol] : 0.0f;
#pragma unroll
            for (int r = 0; r < 8; r++) {
                size_t idx = (size_t)(wm + i * 16 + r + 8 * half) * N + ncol;
                float v = acc[i][j][r] + bv;
                if (NT) __builtin_nontemporal_store(v, &C[idx]);
                else    C[idx] = v;
            }
        }
}

// ---------------------------------------------------------------------------
// K5: GRU scan. 16 persistent blocks; block g owns h columns [g*32, g*32+32).
//     Its 96 needed W_hh rows (r/z/n gate slices) live in LDS for the whole
//     scan. Per step: gh slice via pipelined WMMA (A: global h ping-pong,
//     B: LDS), gate math, publish bf16 h slice, one device-wide barrier.
// ---------------------------------------------------------------------------
__global__ __launch_bounds__(384) void gru_scan(
    const bf16* __restrict__ Whh_b, const float* __restrict__ bhh,
    const float* __restrict__ gi, const float* __restrict__ h0f,
    bf16* __restrict__ hbuf /* 2 x [32][512] */, bf16* __restrict__ hs_b,
    unsigned* __restrict__ bar) {
    __shared__ __align__(32) bf16 whh_s[96 * 512];   // 96 KB: owned W_hh rows
    __shared__ float gh_s[32 * 96];                  // 12 KB
    __shared__ float h_s[32 * 32];                   //  4 KB

    const int g = blockIdx.x;   // 0..15

    // Stage W_hh rows {reg*512 + g*32 + c} -> LDS row reg*32+c.
    for (int idx = threadIdx.x; idx < 96 * 64; idx += blockDim.x) {
        int row = idx >> 6;
        int chunk = idx & 63;
        int reg = row >> 5, c = row & 31;
        const bf16* src = Whh_b + (size_t)(reg * 512 + g * 32 + c) * 512 + chunk * 8;
        *(v8bf*)(whh_s + row * 512 + chunk * 8) = *(const v8bf*)src;
    }
    for (int idx = threadIdx.x; idx < 32 * 32; idx += blockDim.x) {
        int b = idx >> 5, c = idx & 31;
        h_s[idx] = h0f[b * 512 + g * 32 + c];
    }
    __syncthreads();

    const int wave = threadIdx.x >> 5;   // 0..11
    const int wm = (wave & 1) * 16;      // M rows 0..15 or 16..31
    const int jt = wave >> 1;            // N tile 0..5 within 96-col slice
    const int lane = threadIdx.x & 31;
    const int n0 = lane & 15, half = lane >> 4;

    for (int t = 0; t < T_; t++) {
        const bf16* hin = hbuf + (size_t)(t & 1) * (32 * 512);
        bf16* hout      = hbuf + (size_t)((t + 1) & 1) * (32 * 512);

        // gh tile [16x16] at (wm, jt*16), K=512, 2-deep pipelined.
        v8f acc = (v8f){0.f, 0.f, 0.f, 0.f, 0.f, 0.f, 0.f, 0.f};
        v16bf pa0 = load_frag_A(hin, 512, wm, 0);
        v16bf pb0 = load_frag_B(whh_s, 512, jt * 16, 0);
#pragma unroll
        for (int k = 0; k < 512; k += 64) {
            v16bf pa1 = load_frag_A(hin, 512, wm, k + 32);
            v16bf pb1 = load_frag_B(whh_s, 512, jt * 16, k + 32);
            acc = wmma_bf16(pa0, pb0, acc);
            if (k + 64 < 512) {
                pa0 = load_frag_A(hin, 512, wm, k + 64);
                pb0 = load_frag_B(whh_s, 512, jt * 16, k + 64);
            }
            acc = wmma_bf16(pa1, pb1, acc);
        }
#pragma unroll
        for (int r = 0; r < 8; r++)
            gh_s[(wm + r + 8 * half) * 96 + jt * 16 + n0] = acc[r];
        __syncthreads();

        // Gate update for this block's 32x32 slice.
        for (int idx = threadIdx.x; idx < 32 * 32; idx += blockDim.x) {
            int b = idx >> 5, c = idx & 31;
            int col = g * 32 + c;
            int bt = b * T_ + t;
            float hr = gh_s[b * 96 + c]      + bhh[col];
            float hz = gh_s[b * 96 + 32 + c] + bhh[512 + col];
            float hn = gh_s[b * 96 + 64 + c] + bhh[1024 + col];
            const float* gir = gi + (size_t)bt * G3H_;
            float r  = sig_fast(gir[col] + hr);
            float zg = sig_fast(gir[512 + col] + hz);
            float nn = tanh_fast(gir[1024 + col] + r * hn);
            float hnew = (1.0f - zg) * nn + zg * h_s[idx];
            h_s[idx] = hnew;
            bf16 hb = (bf16)hnew;
            hout[b * 512 + col] = hb;
            hs_b[(size_t)bt * 512 + col] = hb;
        }

        // Device-wide barrier before t+1 consumes hout.
        __threadfence();
        __syncthreads();
        if (threadIdx.x == 0) {
            __hip_atomic_fetch_add(bar, 1u, __ATOMIC_RELEASE, __HIP_MEMORY_SCOPE_AGENT);
            unsigned target = (unsigned)(t + 1) * SCAN_BLOCKS;
            while (__hip_atomic_load(bar, __ATOMIC_ACQUIRE, __HIP_MEMORY_SCOPE_AGENT) < target)
                __builtin_amdgcn_s_sleep(2);
        }
        __syncthreads();
    }
}

// ---------------------------------------------------------------------------
// launch
// ---------------------------------------------------------------------------
extern "C" void kernel_launch(void* const* d_in, const int* in_sizes, int n_in,
                              void* d_out, int out_size, void* d_ws, size_t ws_size,
                              hipStream_t stream) {
    const int*   x     = (const int*)d_in[0];
    const float* z     = (const float*)d_in[1];
    const float* emb   = (const float*)d_in[2];
    const float* Winit = (const float*)d_in[3];
    const float* binit = (const float*)d_in[4];
    const float* Wih   = (const float*)d_in[5];
    const float* Whh   = (const float*)d_in[6];
    const float* bih   = (const float*)d_in[7];
    const float* bhh   = (const float*)d_in[8];
    const float* Wout  = (const float*)d_in[9];
    float* out = (float*)d_out;

    char* ws = (char*)d_ws;
    size_t off = 0;
    auto alloc = [&](size_t bytes) -> void* {
        off = (off + 255) & ~(size_t)255;
        void* p = ws + off;
        off += bytes;
        return p;
    };
    bf16*     wout_b = (bf16*)alloc((size_t)V_ * H_ * 2);
    bf16*     wih_b  = (bf16*)alloc((size_t)G3H_ * KIN_ * 2);
    bf16*     whh_b  = (bf16*)alloc((size_t)G3H_ * H_ * 2);
    bf16*     rnin_b = (bf16*)alloc((size_t)M_ * KIN_ * 2);
    float*    gi_f   = (float*)alloc((size_t)M_ * G3H_ * 4);
    float*    h0f    = (float*)alloc((size_t)B_ * H_ * 4);
    bf16*     hbuf   = (bf16*)alloc((size_t)2 * B_ * H_ * 2);
    bf16*     hs_b   = (bf16*)alloc((size_t)M_ * H_ * 2);
    unsigned* bar    = (unsigned*)alloc(256);
    (void)ws_size; (void)in_sizes; (void)n_in; (void)out_size;

    // 1) weight conversions to bf16 (b128 NT loads)
    f32_to_bf16_v4<<<2048, 256, 0, stream>>>(Wout, wout_b, (V_ * H_) / 4);
    f32_to_bf16_v4<<<512,  256, 0, stream>>>(Wih,  wih_b,  (G3H_ * KIN_) / 4);
    f32_to_bf16_v4<<<512,  256, 0, stream>>>(Whh,  whh_b,  (G3H_ * H_) / 4);

    // 2) rnn_in = concat(emb[x], z) as bf16 (8 elems/thread)
    build_rnn_in<<<(M_ * (KIN_ / 8) + 255) / 256, 256, 0, stream>>>(x, z, emb, rnin_b);

    // 3) h0 (also resets scan barrier)
    h0_kernel<<<64, 256, 0, stream>>>(z, Winit, binit, h0f, hbuf, bar);

    // 4) gi = rnn_in @ W_ih^T + b_ih   (M=4096, N=1536, K=768)
    gemm_bf16_nt<KIN_, true, false><<<dim3(G3H_ / 256, M_ / 128), 256, 0, stream>>>(
        rnin_b, wih_b, bih, gi_f, G3H_);

    // 5) GRU scan -> hs (bf16)
    gru_scan<<<SCAN_BLOCKS, 384, 0, stream>>>(whh_b, bhh, gi_f, h0f, hbuf, hs_b, bar);

    // 6) logits = hs @ W_out^T         (M=4096, N=32000, K=512), NT stores
    gemm_bf16_nt<H_, false, true><<<dim3(V_ / 256, M_ / 128), 256, 0, stream>>>(
        hs_b, wout_b, nullptr, out, V_);
}
